// DCGRUEncoder_59270548684895
// MI455X (gfx1250) — matrix operand order repaired and checked by reading, not compiled, not run on hardware
//
#include <hip/hip_runtime.h>

// ---------------------------------------------------------------------------
// DCGRU encoder for MI455X (gfx1250, wave32, WMMA + Tensor Data Mover).
// Heavy math through v_wmma_f32_16x16x32_bf16 (bf16 A/B, f32 accum).
// A-tiles staged via TDM tensor_load_to_lds (async, TENSORcnt) when available.
// Activation layout internally: (n, b, c) so diffusion einsum == plain GEMM.
// ---------------------------------------------------------------------------

typedef __bf16 bf16;
typedef __attribute__((ext_vector_type(16))) __bf16 v16bf;
typedef __attribute__((ext_vector_type(8)))  __bf16 v8bf;
typedef __attribute__((ext_vector_type(8)))  float   v8f;

#define LDSTR 72  // padded LDS row stride in halfs (64 data + 8 pad) -> conflict-free frag reads

__device__ __forceinline__ v16bf cat8(v8bf lo, v8bf hi) {
  return __builtin_shufflevector(lo, hi, 0,1,2,3,4,5,6,7,8,9,10,11,12,13,14,15);
}

__device__ __forceinline__ v8bf zero8() {
  v8bf v;
#pragma unroll
  for (int i = 0; i < 8; ++i) v[i] = (bf16)0.0f;
  return v;
}

// ---------------------------------------------------------------------------
// Tensor Data Mover: pull a 128-row x 64-half 2D tile (row stride = `stride`
// halfs) into LDS at `lds_off`, padding each 128B LDS row with 16B so the
// resulting LDS stride is LDSTR halfs.  D# layout per CDNA5 ISA ch.8:
//   group0: count=1 | lds_addr | global_addr[56:0] | type=2
//   group1: data_size=2B, pad_enable, pad_interval=32dw, pad_amount=4dw,
//           tensor_dim0/1 (for OOB zero-fill), tile 64x128, dim0 stride.
// Issued by one wave; completion via s_wait_tensorcnt before the WG barrier.
// This toolchain's builtin takes 6 args (g0, g1, g2, g3, extra, cpol).
// ---------------------------------------------------------------------------
#if defined(__gfx1250__) && __has_builtin(__builtin_amdgcn_tensor_load_to_lds)
#define USE_TDM 1
typedef __attribute__((ext_vector_type(4))) unsigned int v4u;
typedef __attribute__((ext_vector_type(8))) int          v8i;
typedef __attribute__((ext_vector_type(4))) int          v4i;

__device__ __forceinline__ void tdm_load_a(const bf16* gptr, unsigned lds_off,
                                           unsigned dim0, unsigned rows,
                                           unsigned stride) {
  unsigned long long ga = (unsigned long long)gptr;
  v4u g0;
  g0[0] = 1u;                                    // count=1 (user descriptor)
  g0[1] = lds_off;                               // lds_addr (bytes)
  g0[2] = (unsigned)ga;                          // global_addr[31:0]
  g0[3] = (unsigned)((ga >> 32) & 0x1ffffffu) | (2u << 30);  // addr[56:32] | type=2
  v8i g1;
  g1[0] = (int)((1u << 16) |                     // data_size = 2 bytes
                (1u << 20) |                     // pad_enable
                (4u << 22) |                     // pad_interval: 32 dwords (128B)
                (3u << 25));                     // pad_amount: 4 dwords (16B)
  g1[1] = (int)((dim0 & 0xffffu) << 16);         // tensor_dim0[15:0] (bits 63:48)
  g1[2] = (int)((dim0 >> 16) | ((rows & 0xffffu) << 16));   // dim0[31:16] | dim1[15:0]
  g1[3] = (int)((rows >> 16) | (64u << 16));     // dim1[31:16] | tile_dim0=64
  g1[4] = 128;                                   // tile_dim1=128, tile_dim2=0
  g1[5] = (int)stride;                           // tensor_dim0_stride[31:0]
  g1[6] = 0;                                     // stride0[47:32]=0, stride1[15:0]=0
  g1[7] = 0;
  v4i z4 = {0, 0, 0, 0};
  v8i z8 = {0, 0, 0, 0, 0, 0, 0, 0};
  __builtin_amdgcn_tensor_load_to_lds(g0, g1, z4, z4, z8, 0);
}
#endif

// Shared WMMA inner tile: 64-deep K step from LDS tiles As(row-major) / Bs(transposed).
__device__ __forceinline__ void wmma_tile64(const bf16* As, const bf16* Bs,
                                            int wm0, int wn0, int lhalf, int lcol,
                                            v8f (&acc)[4][2]) {
#pragma unroll
  for (int ks = 0; ks < 64; ks += 32) {
    v16bf af[4], bg[2];
#pragma unroll
    for (int i = 0; i < 4; ++i) {
      // A frag (16x32 bf16): lane holds row M=lane%16; halfs [k0+sel*8,+8) and [k0+16+sel*8,+8)
      const bf16* p = As + (wm0 + i * 16 + lcol) * LDSTR + ks + lhalf * 8;
      af[i] = cat8(*(const v8bf*)p, *(const v8bf*)(p + 16));
    }
#pragma unroll
    for (int j = 0; j < 2; ++j) {
      // B frag (32x16 bf16): lane holds col N=lane%16; halfs [k0+sel*8,+8) and [k0+16+sel*8,+8)
      const bf16* p = Bs + (wn0 + j * 16 + lcol) * LDSTR + ks + lhalf * 8;
      bg[j] = cat8(*(const v8bf*)p, *(const v8bf*)(p + 16));
    }
#pragma unroll
    for (int i = 0; i < 4; ++i)
#pragma unroll
      for (int j = 0; j < 2; ++j)
        acc[i][j] = __builtin_amdgcn_wmma_f32_16x16x32_bf16(
            false, af[i], false, bg[j], (short)0, acc[i][j], false, false);
  }
}

// ---------------------------------------------------------------------------
// Diffusion GEMM:  Zout[term dst] = T(512x512) @ Zin[term src]
// Logical B/out matrix is (512 x 32*C); column j maps to z[.][j/C][off + j%C].
// z row stride = 32*d (d = 7*C). C is a multiple of 16 so 16-wide WMMA column
// tiles and 8-wide load chunks never straddle a batch boundary.
// ---------------------------------------------------------------------------
__global__ __launch_bounds__(256)
void gemm_diff_kernel(const bf16* __restrict__ T,
                      const bf16* __restrict__ Zin,
                      bf16* __restrict__ Zout,
                      int C, int d, int srcOff, int dstOff) {
  __shared__ bf16 As[128 * LDSTR];
  __shared__ bf16 Bs[128 * LDSTR];  // transposed: [col][k]

  const int tid  = threadIdx.x;
  const int m0   = blockIdx.y * 128;
  const int j0   = blockIdx.x * 128;
  const long rs  = 32L * d;  // z row stride (elements)

  const int wid   = tid >> 5;
  const int lane  = tid & 31;
  const int wm0   = (wid & 1) * 64;
  const int wn0   = (wid >> 1) * 32;
  const int lhalf = lane >> 4;
  const int lcol  = lane & 15;

  v8f acc[4][2];
#pragma unroll
  for (int i = 0; i < 4; ++i)
#pragma unroll
    for (int j = 0; j < 2; ++j)
#pragma unroll
      for (int e = 0; e < 8; ++e) acc[i][j][e] = 0.0f;

  for (int k0 = 0; k0 < 512; k0 += 64) {
#ifdef USE_TDM
    // A tile via Tensor Data Mover (wave 0 issues; others proceed to B tile)
    if (wid == 0) {
      tdm_load_a(T + (long)m0 * 512 + k0,
                 (unsigned)(unsigned long long)(&As[0]),
                 (unsigned)(512 - k0), (unsigned)(512 - m0), 512u);
    }
#else
    // A tile: 128x64 from T (row-major, lda=512); 4 x v8bf per thread
#pragma unroll
    for (int q = 0; q < 4; ++q) {
      int cid = tid + q * 256;
      int row = cid >> 3;
      int kc  = (cid & 7) * 8;
      v8bf v = *(const v8bf*)(T + (long)(m0 + row) * 512 + k0 + kc);
      *(v8bf*)(&As[row * LDSTR + kc]) = v;
    }
#endif
    // B tile: 64(k) x 128(col), transposed into Bs[col][k]
#pragma unroll
    for (int q = 0; q < 4; ++q) {
      int cid  = tid + q * 256;
      int krow = cid >> 4;
      int col0 = (cid & 15) * 8;
      int jg   = j0 + col0;
      int bidx = jg / C;
      long a   = (long)(k0 + krow) * rs + (long)bidx * d + srcOff + (jg - bidx * C);
      v8bf v = *(const v8bf*)(Zin + a);
#pragma unroll
      for (int i = 0; i < 8; ++i) Bs[(col0 + i) * LDSTR + krow] = v[i];
    }
#ifdef USE_TDM
    if (wid == 0) __builtin_amdgcn_s_wait_tensorcnt(0);
#endif
    __syncthreads();
    wmma_tile64(As, Bs, wm0, wn0, lhalf, lcol, acc);
    __syncthreads();
  }

  // Store bf16 into z at term dstOff. C/D frag: row=(lane/16)*8+e, col=lane%16.
#pragma unroll
  for (int j = 0; j < 2; ++j) {
    int jg0   = j0 + wn0 + j * 16;
    int bidx  = jg0 / C;
    long colB = (long)bidx * d + dstOff + (jg0 - bidx * C) + lcol;
#pragma unroll
    for (int i = 0; i < 4; ++i) {
      int mB = m0 + wm0 + i * 16 + lhalf * 8;
#pragma unroll
      for (int e = 0; e < 8; ++e)
        Zout[(long)(mB + e) * rs + colB] = (bf16)acc[i][j][e];
    }
  }
}

// ---------------------------------------------------------------------------
// Projection GEMM:  Out(16384 x no) = act( Z(16384 x Kd) @ W(Kd x no) + bias )
// act: 0 = none, 1 = sigmoid, 2 = tanh.  Kd may not be a multiple of 64
// (d0 = 1008): TDM zero-fills reads past tensor_dim0; fallback zero-fills LDS.
// ---------------------------------------------------------------------------
__global__ __launch_bounds__(256)
void gemm_proj_kernel(const bf16* __restrict__ Z, int Kd,
                      const bf16* __restrict__ W, int no,
                      const float* __restrict__ bias,
                      float* __restrict__ Out, int act) {
  __shared__ bf16 As[128 * LDSTR];
  __shared__ bf16 Bs[128 * LDSTR];

  const int tid = threadIdx.x;
  const int m0  = blockIdx.y * 128;
  const int j0  = blockIdx.x * 128;

  const int wid   = tid >> 5;
  const int lane  = tid & 31;
  const int wm0   = (wid & 1) * 64;
  const int wn0   = (wid >> 1) * 32;
  const int lhalf = lane >> 4;
  const int lcol  = lane & 15;

  v8f acc[4][2];
#pragma unroll
  for (int i = 0; i < 4; ++i)
#pragma unroll
    for (int j = 0; j < 2; ++j)
#pragma unroll
      for (int e = 0; e < 8; ++e) acc[i][j][e] = 0.0f;

  const int kTiles = (Kd + 63) >> 6;
  for (int kt = 0; kt < kTiles; ++kt) {
    const int k0 = kt * 64;
#ifdef USE_TDM
    if (wid == 0) {
      tdm_load_a(Z + (long)m0 * Kd + k0,
                 (unsigned)(unsigned long long)(&As[0]),
                 (unsigned)(Kd - k0), (unsigned)(16384 - m0), (unsigned)Kd);
    }
#else
#pragma unroll
    for (int q = 0; q < 4; ++q) {
      int cid = tid + q * 256;
      int row = cid >> 3;
      int kc  = (cid & 7) * 8;
      v8bf v = (k0 + kc < Kd) ? *(const v8bf*)(Z + (long)(m0 + row) * Kd + k0 + kc)
                              : zero8();
      *(v8bf*)(&As[row * LDSTR + kc]) = v;
    }
#endif
#pragma unroll
    for (int q = 0; q < 4; ++q) {
      int cid  = tid + q * 256;
      int krow = cid >> 4;
      int col0 = (cid & 15) * 8;
      int k    = k0 + krow;
      v8bf v = (k < Kd) ? *(const v8bf*)(W + (long)k * no + j0 + col0) : zero8();
#pragma unroll
      for (int i = 0; i < 8; ++i) Bs[(col0 + i) * LDSTR + krow] = v[i];
    }
#ifdef USE_TDM
    if (wid == 0) __builtin_amdgcn_s_wait_tensorcnt(0);
#endif
    __syncthreads();
    wmma_tile64(As, Bs, wm0, wn0, lhalf, lcol, acc);
    __syncthreads();
  }

#pragma unroll
  for (int j = 0; j < 2; ++j) {
    int jg = j0 + wn0 + j * 16 + lcol;
    float bs = bias[jg];
#pragma unroll
    for (int i = 0; i < 4; ++i) {
      int mB = m0 + wm0 + i * 16 + lhalf * 8;
#pragma unroll
      for (int e = 0; e < 8; ++e) {
        float v = acc[i][j][e] + bs;
        if (act == 1)      v = 1.0f / (1.0f + expf(-v));
        else if (act == 2) v = tanhf(v);
        Out[(long)(mB + e) * no + jg] = v;
      }
    }
  }
}

// ---------------------------------------------------------------------------
// Elementwise helpers
// ---------------------------------------------------------------------------
__global__ void cvt_kernel(const float* __restrict__ s, bf16* __restrict__ d, long n) {
  long i = (long)blockIdx.x * 256 + threadIdx.x;
  if (i < n) d[i] = (bf16)s[i];
}

__global__ void zero_kernel(float* __restrict__ p, long n) {
  long i = (long)blockIdx.x * 256 + threadIdx.x;
  if (i < n) p[i] = 0.0f;
}

// Layer 0 concat: z[n][b][0:16] = x_seq[b][t][n][:],  z[n][b][16:144] = h0[n][b][:]
__global__ void concat0_kernel(const float* __restrict__ x_seq, int t,
                               const float* __restrict__ h0, bf16* __restrict__ z) {
  long id = (long)blockIdx.x * 256 + threadIdx.x;
  const long total = 512L * 32 * 144;
  if (id >= total) return;
  int c = (int)(id % 144);
  long r = id / 144;            // r = n*32 + b
  int b = (int)(r % 32);
  int n = (int)(r / 32);
  float v;
  if (c < 16) v = x_seq[(((long)b * 8 + t) * 512 + n) * 16 + c];
  else        v = h0[r * 128 + (c - 16)];
  z[r * 1008 + c] = (bf16)v;
}

// Layer 1 concat: z[n][b][0:128] = h0,  z[n][b][128:256] = h1
__global__ void concat1_kernel(const float* __restrict__ h0,
                               const float* __restrict__ h1, bf16* __restrict__ z) {
  long id = (long)blockIdx.x * 256 + threadIdx.x;
  const long total = 512L * 32 * 256;
  if (id >= total) return;
  int c = (int)(id & 255);
  long r = id >> 8;
  float v = (c < 128) ? h0[r * 128 + c] : h1[r * 128 + (c - 128)];
  z[r * 1792 + c] = (bf16)v;
}

// Overwrite h-part of z term 0 with r*h  (r = gates[..., :128])
__global__ void apply_r_kernel(const float* __restrict__ gates,
                               const float* __restrict__ h,
                               bf16* __restrict__ z, int xdim, int d) {
  long id = (long)blockIdx.x * 256 + threadIdx.x;
  const long total = 512L * 32 * 128;
  if (id >= total) return;
  int j = (int)(id & 127);
  long r = id >> 7;
  float rv = gates[r * 256 + j];
  z[r * (long)d + xdim + j] = (bf16)(rv * h[r * 128 + j]);
}

// h = LN( (1-u)*h + u*c ), one block (128 threads) per (n,b) row
__global__ __launch_bounds__(128)
void update_ln_kernel(const float* __restrict__ gates, const float* __restrict__ cand,
                      const float* __restrict__ gamma, const float* __restrict__ beta,
                      float* __restrict__ h) {
  __shared__ float s1[128], s2[128];
  long r = blockIdx.x;
  int  j = threadIdx.x;
  float u  = gates[r * 256 + 128 + j];
  float cv = cand[r * 128 + j];
  float hn = (1.0f - u) * h[r * 128 + j] + u * cv;
  s1[j] = hn; s2[j] = hn * hn;
  __syncthreads();
  for (int s = 64; s > 0; s >>= 1) {
    if (j < s) { s1[j] += s1[j + s]; s2[j] += s2[j + s]; }
    __syncthreads();
  }
  float mu  = s1[0] * (1.0f / 128.0f);
  float var = s2[0] * (1.0f / 128.0f) - mu * mu;
  h[r * 128 + j] = (hn - mu) * rsqrtf(var + 1e-5f) * gamma[j] + beta[j];
}

// (n,b,H) -> (B,N,H) for both outputs
__global__ void writeout_kernel(const float* __restrict__ h0,
                                const float* __restrict__ h1,
                                float* __restrict__ out) {
  long id = (long)blockIdx.x * 256 + threadIdx.x;
  const long half = 32L * 512 * 128;
  if (id >= 2 * half) return;
  const float* src = (id < half) ? h0 : h1;
  long k = (id < half) ? id : id - half;
  int j = (int)(k & 127);
  long r = k >> 7;
  int n = (int)(r % 512);
  int b = (int)(r / 512);
  out[id] = src[((long)n * 32 + b) * 128 + j];
}

// ---------------------------------------------------------------------------
extern "C" void kernel_launch(void* const* d_in, const int* in_sizes, int n_in,
                              void* d_out, int out_size, void* d_ws, size_t ws_size,
                              hipStream_t stream) {
  (void)in_sizes; (void)n_in; (void)out_size; (void)ws_size;

  const float* x_seq = (const float*)d_in[0];
  const float* T_f   = (const float*)d_in[1];
  const float* T_b   = (const float*)d_in[2];
  const float* Wg0   = (const float*)d_in[3];
  const float* bg0   = (const float*)d_in[4];
  const float* Wc0   = (const float*)d_in[5];
  const float* bc0   = (const float*)d_in[6];
  const float* g0    = (const float*)d_in[7];
  const float* be0   = (const float*)d_in[8];
  const float* Wg1   = (const float*)d_in[9];
  const float* bg1   = (const float*)d_in[10];
  const float* Wc1   = (const float*)d_in[11];
  const float* bc1   = (const float*)d_in[12];
  const float* g1    = (const float*)d_in[13];
  const float* be1   = (const float*)d_in[14];

  const int C0 = 144, D0 = 1008, C1 = 256, D1 = 1792;

  // Workspace carve-up (all offsets 256B aligned)
  char* ws = (char*)d_ws;
  size_t off = 0;
  auto carve = [&](size_t bytes) { char* p = ws + off; off += (bytes + 255) & ~(size_t)255; return p; };
  bf16*  Tf_bf  = (bf16*)carve(512L * 512 * 2);
  bf16*  Tb_bf  = (bf16*)carve(512L * 512 * 2);
  bf16*  Wg0_bf = (bf16*)carve((long)D0 * 256 * 2);
  bf16*  Wc0_bf = (bf16*)carve((long)D0 * 128 * 2);
  bf16*  Wg1_bf = (bf16*)carve((long)D1 * 256 * 2);
  bf16*  Wc1_bf = (bf16*)carve((long)D1 * 128 * 2);
  bf16*  zb     = (bf16*)carve(512L * 32 * D1 * 2);
  float* h0     = (float*)carve(512L * 32 * 128 * 4);
  float* h1     = (float*)carve(512L * 32 * 128 * 4);
  float* gates  = (float*)carve(512L * 32 * 256 * 4);
  float* cand   = (float*)carve(512L * 32 * 128 * 4);

  auto gsz = [](long n) { return dim3((unsigned)((n + 255) / 256)); };

  // One-time (per launch) conversions / init
  cvt_kernel<<<gsz(512L * 512), 256, 0, stream>>>(T_f, Tf_bf, 512L * 512);
  cvt_kernel<<<gsz(512L * 512), 256, 0, stream>>>(T_b, Tb_bf, 512L * 512);
  cvt_kernel<<<gsz((long)D0 * 256), 256, 0, stream>>>(Wg0, Wg0_bf, (long)D0 * 256);
  cvt_kernel<<<gsz((long)D0 * 128), 256, 0, stream>>>(Wc0, Wc0_bf, (long)D0 * 128);
  cvt_kernel<<<gsz((long)D1 * 256), 256, 0, stream>>>(Wg1, Wg1_bf, (long)D1 * 256);
  cvt_kernel<<<gsz((long)D1 * 128), 256, 0, stream>>>(Wc1, Wc1_bf, (long)D1 * 128);
  zero_kernel<<<gsz(512L * 32 * 128), 256, 0, stream>>>(h0, 512L * 32 * 128);
  zero_kernel<<<gsz(512L * 32 * 128), 256, 0, stream>>>(h1, 512L * 32 * 128);

  auto diff6 = [&](const bf16* Tfm, const bf16* Tbm, int C, int d) {
    dim3 g((unsigned)((32 * C) / 128), 4);
    gemm_diff_kernel<<<g, 256, 0, stream>>>(Tfm, zb, zb, C, d, 0 * C, 1 * C);
    gemm_diff_kernel<<<g, 256, 0, stream>>>(Tfm, zb, zb, C, d, 1 * C, 2 * C);
    gemm_diff_kernel<<<g, 256, 0, stream>>>(Tfm, zb, zb, C, d, 2 * C, 3 * C);
    gemm_diff_kernel<<<g, 256, 0, stream>>>(Tbm, zb, zb, C, d, 0 * C, 4 * C);
    gemm_diff_kernel<<<g, 256, 0, stream>>>(Tbm, zb, zb, C, d, 4 * C, 5 * C);
    gemm_diff_kernel<<<g, 256, 0, stream>>>(Tbm, zb, zb, C, d, 5 * C, 6 * C);
  };

  const long NB  = 512L * 32;          // rows for elementwise per-(n,b) work
  const dim3 gProjG(2, 128);           // no = 256
  const dim3 gProjC(1, 128);           // no = 128

  for (int t = 0; t < 8; ++t) {
    // ---------------- layer 0: x = x_seq[:,t], h = h0 ----------------
    concat0_kernel<<<gsz(NB * C0), 256, 0, stream>>>(x_seq, t, h0, zb);
    diff6(Tf_bf, Tb_bf, C0, D0);
    gemm_proj_kernel<<<gProjG, 256, 0, stream>>>(zb, D0, Wg0_bf, 256, bg0, gates, 1);
    apply_r_kernel<<<gsz(NB * 128), 256, 0, stream>>>(gates, h0, zb, 16, D0);
    diff6(Tf_bf, Tb_bf, C0, D0);
    gemm_proj_kernel<<<gProjC, 256, 0, stream>>>(zb, D0, Wc0_bf, 128, bc0, cand, 2);
    update_ln_kernel<<<dim3(16384), 128, 0, stream>>>(gates, cand, g0, be0, h0);

    // ---------------- layer 1: x = h0, h = h1 ----------------
    concat1_kernel<<<gsz(NB * C1), 256, 0, stream>>>(h0, h1, zb);
    diff6(Tf_bf, Tb_bf, C1, D1);
    gemm_proj_kernel<<<gProjG, 256, 0, stream>>>(zb, D1, Wg1_bf, 256, bg1, gates, 1);
    apply_r_kernel<<<gsz(NB * 128), 256, 0, stream>>>(gates, h1, zb, 128, D1);
    diff6(Tf_bf, Tb_bf, C1, D1);
    gemm_proj_kernel<<<gProjC, 256, 0, stream>>>(zb, D1, Wc1_bf, 128, bc1, cand, 2);
    update_ln_kernel<<<dim3(16384), 128, 0, stream>>>(gates, cand, g1, be1, h1);
  }

  writeout_kernel<<<gsz(2 * NB * 128), 256, 0, stream>>>(h0, h1, (float*)d_out);
}
